// GRU_13030930776564
// MI455X (gfx1250) — matrix operand order repaired and checked by reading, not compiled, not run on hardware
//
#include <hip/hip_runtime.h>
#include <hip/hip_bf16.h>

typedef __attribute__((ext_vector_type(16))) _Float16 v16h;
typedef __attribute__((ext_vector_type(8)))  _Float16 v8h;
typedef __attribute__((ext_vector_type(8)))  float    v8f;
typedef __attribute__((ext_vector_type(4)))  unsigned int u32x4;
typedef __attribute__((ext_vector_type(8)))  int      i32x8;
typedef __attribute__((ext_vector_type(4)))  int      i32x4;

#define B_  64
#define L_  1024
#define D_  256
#define MROWS (B_ * L_)          // 65536
#define MTILES (MROWS / 16)      // 4096
#define LDA 264                  // f16 LDS row stride: 528B (4-bank shift/row)
#define LDF 260                  // f32 LDS row stride: 1040B (4-bank shift/row)

#if defined(__has_builtin)
#  if __has_builtin(__builtin_amdgcn_tensor_load_to_lds)
#    define HAVE_TDM 1
#  endif
#endif
#ifndef HAVE_TDM
#  define HAVE_TDM 0
#endif

// ---------------- WMMA helpers ----------------

__device__ __forceinline__ v8f wmma_f16(v16h a, v16h b, v8f c) {
    return __builtin_amdgcn_wmma_f32_16x16x32_f16(
        false, a, false, b, (short)0, c, false, false);
}

__device__ __forceinline__ v16h pack_a(v8h lo, v8h hi) {
    v16h a;
#pragma unroll
    for (int e = 0; e < 8; ++e) { a[e] = lo[e]; a[e + 8] = hi[e]; }
    return a;
}

// A fragment (16x32 f16) from an LDS row-major tile [16][LDA].
__device__ __forceinline__ v16h load_a_frag(const _Float16* arow, int k0) {
    return pack_a(*(const v8h*)(arow + k0), *(const v8h*)(arow + k0 + 16));
}

// A fragment = sum of two f16 tiles (v_pk_add_f16).
__device__ __forceinline__ v16h load_a_frag2(const _Float16* a1, const _Float16* a2, int k0) {
    v8h lo = *(const v8h*)(a1 + k0)      + *(const v8h*)(a2 + k0);
    v8h hi = *(const v8h*)(a1 + k0 + 16) + *(const v8h*)(a2 + k0 + 16);
    return pack_a(lo, hi);
}

// B fragment (32x16 f16) gathered from global f16 weight W[k][n] (row-major, ld=256).
__device__ __forceinline__ v16h load_b_frag(const _Float16* W, int kk, int half, int n) {
    v16h b;
    const int kbase = kk * 32 + 8 * half;
#pragma unroll
    for (int e = 0; e < 16; ++e) b[e] = W[(kbase + e + (e >= 8 ? 8 : 0)) * D_ + n];
    return b;
}

// ---------------- TDM: DMA one 16x256 tile -> padded LDS ----------------
// dsz_code: 1 = 2B elems (pad_code 6: pad after 128 DWORDs), 2 = 4B elems (pad_code 7).
// Pad amount is always 4 DWORDs (16B) -> LDS strides LDA / LDF.

#if HAVE_TDM
__device__ __forceinline__ void tdm_load_tile(const void* gsrc, unsigned lds_addr,
                                              unsigned dsz_code, unsigned pad_code,
                                              unsigned long long stride_elems) {
    const unsigned long long ga = (unsigned long long)(__SIZE_TYPE__)gsrc;
    u32x4 g0;
    g0[0] = 1u;                                                   // count=1, user desc
    g0[1] = lds_addr;                                             // LDS byte address
    g0[2] = (unsigned)ga;                                         // global_addr[31:0]
    g0[3] = (unsigned)((ga >> 32) & 0x01FFFFFFull) | (2u << 30);  // addr[56:32] | type=2
    i32x8 g1;
    g1[0] = (int)((dsz_code << 16) | (1u << 20) | (pad_code << 22) | (3u << 25));
    g1[1] = (int)(256u << 16);          // tensor_dim0 = 256
    g1[2] = (int)(16u << 16);           // tensor_dim1 = 16
    g1[3] = (int)(256u << 16);          // tile_dim0 = 256
    g1[4] = 16;                         // tile_dim1 = 16, tile_dim2 = 0
    g1[5] = (int)(unsigned)stride_elems;        // tensor_dim0_stride lo32
    g1[6] = (int)(unsigned)(stride_elems >> 32);// stride hi16 (bits 207:192)
    g1[7] = 0;
    const i32x4 z4 = {0, 0, 0, 0};
#if __clang_major__ >= 23
    const i32x8 z8 = {0, 0, 0, 0, 0, 0, 0, 0};
    __builtin_amdgcn_tensor_load_to_lds(g0, g1, z4, z4, z8, 0);
#else
    __builtin_amdgcn_tensor_load_to_lds(g0, g1, z4, z4, 0);
#endif
}
#endif

// ---------------- weight conversion ----------------

__global__ void cvt_f32_f16_kernel(const float* __restrict__ src,
                                   _Float16* __restrict__ dst, int n) {
    int i = blockIdx.x * blockDim.x + threadIdx.x;
    if (i < n) dst[i] = (_Float16)src[i];
}

// ---------------- generic WMMA GEMM: Out[M,256] = A[M,256] @ W[256,256] + bias ----------------
enum { AM_F32 = 0, AM_F16 = 1, AM_SUM = 2 };
enum { OM_F32 = 0, OM_F16 = 1 };

template <int AMODE, int OMODE>
__global__ __launch_bounds__(512)
void gemm256_kernel(const void* __restrict__ Aptr, const void* __restrict__ A2ptr,
                    const _Float16* __restrict__ W, const float* __restrict__ bias,
                    void* __restrict__ Out) {
    __shared__ _Float16 abuf[2][16 * LDA + 8];
    __shared__ _Float16 abuf2[2][16 * LDA + 8];   // second operand for AM_SUM

    const int tid  = threadIdx.x;
    const int lane = tid & 31;
    const int wv   = tid >> 5;
    const int half = lane >> 4;
    const int nl   = lane & 15;
    const int n    = wv * 16 + nl;

    v16h Bf[8];
#pragma unroll
    for (int kk = 0; kk < 8; ++kk) Bf[kk] = load_b_frag(W, kk, half, n);
    const float bv = bias[n];

#if HAVE_TDM
    if (AMODE == AM_F16 || AMODE == AM_SUM) {
        // ---- TDM double-buffered pipeline: DMA tile n+1 while computing tile n ----
        const _Float16* A16 = (const _Float16*)Aptr;
        const _Float16* A26 = (const _Float16*)A2ptr;
        int buf = 0;
        const int mt0 = blockIdx.x;
        if (wv == 0 && mt0 < MTILES) {
            tdm_load_tile(A16 + (long)mt0 * 16 * D_,
                          (unsigned)(__SIZE_TYPE__)&abuf[0][0], 1, 6, 256);
            if (AMODE == AM_SUM)
                tdm_load_tile(A26 + (long)mt0 * 16 * D_,
                              (unsigned)(__SIZE_TYPE__)&abuf2[0][0], 1, 6, 256);
        }
        __builtin_amdgcn_s_wait_tensorcnt(0);
        __syncthreads();

        for (int mt = mt0; mt < MTILES; mt += gridDim.x) {
            const int mnext = mt + gridDim.x;
            if (wv == 0 && mnext < MTILES) {
                tdm_load_tile(A16 + (long)mnext * 16 * D_,
                              (unsigned)(__SIZE_TYPE__)&abuf[buf ^ 1][0], 1, 6, 256);
                if (AMODE == AM_SUM)
                    tdm_load_tile(A26 + (long)mnext * 16 * D_,
                                  (unsigned)(__SIZE_TYPE__)&abuf2[buf ^ 1][0], 1, 6, 256);
            }

            const _Float16* arow  = &abuf[buf][nl * LDA];
            const _Float16* arow2 = &abuf2[buf][nl * LDA];
            v8f acc;
#pragma unroll
            for (int i = 0; i < 8; ++i) acc[i] = bv;
#pragma unroll
            for (int kk = 0; kk < 8; ++kk) {
                v16h a = (AMODE == AM_SUM)
                           ? load_a_frag2(arow, arow2, kk * 32 + 8 * half)
                           : load_a_frag(arow, kk * 32 + 8 * half);
                acc = wmma_f16(a, Bf[kk], acc);
            }
            const long m0 = (long)mt * 16;
            if (OMODE == OM_F32) {
                float* O = (float*)Out;
#pragma unroll
                for (int i = 0; i < 8; ++i) O[(m0 + 8 * half + i) * D_ + n] = acc[i];
            } else {
                _Float16* O = (_Float16*)Out;
#pragma unroll
                for (int i = 0; i < 8; ++i) O[(m0 + 8 * half + i) * D_ + n] = (_Float16)acc[i];
            }
            __builtin_amdgcn_s_wait_tensorcnt(0);   // next buffer's DMA complete
            __syncthreads();
            buf ^= 1;
        }
        return;
    }
#endif

    // ---- manual staging path ----
    const int e0   = tid * 8;
    const int srow = e0 >> 8;
    const int scol = e0 & 255;
    _Float16* sdst = &abuf[0][srow * LDA + scol];
    const _Float16* arow = &abuf[0][nl * LDA];

    for (int mt = blockIdx.x; mt < MTILES; mt += gridDim.x) {
        const long m0 = (long)mt * 16;
        if (AMODE == AM_F32) {
            const float* A = (const float*)Aptr + (m0 + srow) * D_ + scol;
#pragma unroll
            for (int e = 0; e < 8; ++e) sdst[e] = (_Float16)A[e];
        } else if (AMODE == AM_F16) {
            const v8h* A = (const v8h*)((const _Float16*)Aptr + (m0 + srow) * D_ + scol);
            *(v8h*)sdst = *A;
        } else {
            const _Float16* A  = (const _Float16*)Aptr  + (m0 + srow) * D_ + scol;
            const _Float16* A2 = (const _Float16*)A2ptr + (m0 + srow) * D_ + scol;
#pragma unroll
            for (int e = 0; e < 8; ++e) sdst[e] = A[e] + A2[e];
        }
        __syncthreads();

        v8f acc;
#pragma unroll
        for (int i = 0; i < 8; ++i) acc[i] = bv;
#pragma unroll
        for (int kk = 0; kk < 8; ++kk) {
            v16h a = load_a_frag(arow, kk * 32 + 8 * half);
            acc = wmma_f16(a, Bf[kk], acc);
        }

        if (OMODE == OM_F32) {
            float* O = (float*)Out;
#pragma unroll
            for (int i = 0; i < 8; ++i) O[(m0 + 8 * half + i) * D_ + n] = acc[i];
        } else {
            _Float16* O = (_Float16*)Out;
#pragma unroll
            for (int i = 0; i < 8; ++i) O[(m0 + 8 * half + i) * D_ + n] = (_Float16)acc[i];
        }
        __syncthreads();
    }
}

// ---------------- persistent GRU scan ----------------
// Grid: 8 WGs = 4 batch tiles x 2 directions. Block 512 = 16 waves.
// Wr/Wu fragments register-resident; h ping-pongs in LDS; gate inputs for step
// t+1 are TDM-DMA'd into LDS while step t computes (double-buffered, 1 barrier/step).
// All LDS addressing is single-base + integer offset so clang infers addrspace(3)
// (pointer-array selects degrade ds_load -> flat_load).
#define HTILE (16 * LDA)                      // f16 elements per h buffer
#define GTILE (16 * LDF)                      // f32 elements per gate tile
#define SCAN_LDS_BYTES (2 * HTILE * 2 + 6 * GTILE * 4)

__global__ __launch_bounds__(512)
void gru_scan_kernel(const float* __restrict__ Xx, const float* __restrict__ Xr,
                     const float* __restrict__ Xu,
                     const _Float16* __restrict__ Wr16, const _Float16* __restrict__ Wu16,
                     const float* __restrict__ Wr_b, const float* __restrict__ Wu_b,
                     _Float16* __restrict__ Hf, _Float16* __restrict__ Hb) {
    extern __shared__ char smem[];
    _Float16* const hb = (_Float16*)smem;                          // [2][HTILE]
    float* const gb = (float*)(smem + 2 * HTILE * sizeof(_Float16)); // [2][3][GTILE]

    const int tid  = threadIdx.x;
    const int lane = tid & 31;
    const int wv   = tid >> 5;
    const int half = lane >> 4;
    const int nl   = lane & 15;
    const int n    = wv * 16 + nl;
    const int dir  = blockIdx.x & 1;
    const int b0   = (blockIdx.x >> 1) * 16;
    _Float16* __restrict__ Hout = dir ? Hb : Hf;

    v16h Br[8], Bu[8];
#pragma unroll
    for (int kk = 0; kk < 8; ++kk) {
        Br[kk] = load_b_frag(Wr16, kk, half, n);
        Bu[kk] = load_b_frag(Wu16, kk, half, n);
    }
    const float br = Wr_b[n];
    const float bu = Wu_b[n];

    for (int i = tid; i < HTILE; i += 512) hb[i] = (_Float16)0.f;

#if HAVE_TDM
    if (wv == 0) {   // DMA gate tiles for t=0 into buffer 0
        const int tt0 = dir ? (L_ - 1) : 0;
        const long goff = ((long)b0 * L_ + tt0) * D_;
        const unsigned long long gstride = (unsigned long long)L_ * D_;
        tdm_load_tile(Xx + goff, (unsigned)(__SIZE_TYPE__)(gb + 0 * GTILE), 2, 7, gstride);
        tdm_load_tile(Xr + goff, (unsigned)(__SIZE_TYPE__)(gb + 1 * GTILE), 2, 7, gstride);
        tdm_load_tile(Xu + goff, (unsigned)(__SIZE_TYPE__)(gb + 2 * GTILE), 2, 7, gstride);
    }
    __builtin_amdgcn_s_wait_tensorcnt(0);
#endif
    __syncthreads();

    v8f h;
#pragma unroll
    for (int i = 0; i < 8; ++i) h[i] = 0.f;

    for (int t = 0; t < L_; ++t) {
        const int cur = t & 1;
        const int tt  = dir ? (L_ - 1 - t) : t;
        const int hoff_cur = cur * HTILE;          // integer offsets, single LDS base
        const int hoff_nxt = (cur ^ 1) * HTILE;
        const int goff_cur = cur * (3 * GTILE);

#if HAVE_TDM
        if (wv == 0 && t + 1 < L_) {   // DMA next step's gate tiles (overlaps compute)
            const int t2 = dir ? (L_ - 2 - t) : (t + 1);
            const long goff = ((long)b0 * L_ + t2) * D_;
            const unsigned long long gstride = (unsigned long long)L_ * D_;
            float* gnx = gb + (cur ^ 1) * (3 * GTILE);
            tdm_load_tile(Xx + goff, (unsigned)(__SIZE_TYPE__)(gnx + 0 * GTILE), 2, 7, gstride);
            tdm_load_tile(Xr + goff, (unsigned)(__SIZE_TYPE__)(gnx + 1 * GTILE), 2, 7, gstride);
            tdm_load_tile(Xu + goff, (unsigned)(__SIZE_TYPE__)(gnx + 2 * GTILE), 2, 7, gstride);
        }
        v8f xx, xr, xu;
#pragma unroll
        for (int i = 0; i < 8; ++i) {
            const int mo = goff_cur + (8 * half + i) * LDF + n;
            xx[i] = gb[mo];
            xr[i] = gb[mo + GTILE];
            xu[i] = gb[mo + 2 * GTILE];
        }
#else
        v8f xx, xr, xu;
#pragma unroll
        for (int i = 0; i < 8; ++i) {
            const long idx = ((long)(b0 + 8 * half + i) * L_ + tt) * D_ + n;
            xr[i] = Xr[idx]; xu[i] = Xu[idx]; xx[i] = Xx[idx];
        }
#endif

        v8f ar, au;
#pragma unroll
        for (int i = 0; i < 8; ++i) { ar[i] = br; au[i] = bu; }

        const _Float16* arow = hb + hoff_cur + nl * LDA;
#pragma unroll
        for (int kk = 0; kk < 8; ++kk) {
            v16h a = load_a_frag(arow, kk * 32 + 8 * half);
            ar = wmma_f16(a, Br[kk], ar);
            au = wmma_f16(a, Bu[kk], au);
        }

        // r=sigmoid(ar+xr), u=sigmoid(au+xu), cand=tanh(r*h+xx)  (v_rcp_f32, no div)
#pragma unroll
        for (int i = 0; i < 8; ++i) {
            const float r = __builtin_amdgcn_rcpf(1.f + __expf(-(ar[i] + xr[i])));
            const float u = __builtin_amdgcn_rcpf(1.f + __expf(-(au[i] + xu[i])));
            const float z = r * h[i] + xx[i];
            const float e = __expf(-2.f * fabsf(z));
            const float th = (1.f - e) * __builtin_amdgcn_rcpf(1.f + e);
            const float cand = (z >= 0.f) ? th : -th;
            h[i] = u * h[i] + (1.f - u) * cand;
        }

#pragma unroll
        for (int i = 0; i < 8; ++i) {
            const long idx = ((long)(b0 + 8 * half + i) * L_ + tt) * D_ + n;
            Hout[idx] = (_Float16)h[i];
        }

        // publish new h into the other buffer; single barrier per step
#pragma unroll
        for (int i = 0; i < 8; ++i)
            hb[hoff_nxt + (8 * half + i) * LDA + n] = (_Float16)h[i];

#if HAVE_TDM
        __builtin_amdgcn_s_wait_tensorcnt(0);   // next step's gate DMAs complete
#endif
        __syncthreads();
    }
}

// ---------------- host launch ----------------

extern "C" void kernel_launch(void* const* d_in, const int* in_sizes, int n_in,
                              void* d_out, int out_size, void* d_ws, size_t ws_size,
                              hipStream_t stream) {
    const float* inputs = (const float*)d_in[0];
    const float* Wsrc[7] = {
        (const float*)d_in[1],  // Wi_w
        (const float*)d_in[3],  // Wx_w
        (const float*)d_in[5],  // Ux_w
        (const float*)d_in[7],  // Rx_w
        (const float*)d_in[9],  // Wr_w
        (const float*)d_in[11], // Wu_w
        (const float*)d_in[13]  // Wo_w
    };
    const float* Wi_b = (const float*)d_in[2];
    const float* Wx_b = (const float*)d_in[4];
    const float* Ux_b = (const float*)d_in[6];
    const float* Rx_b = (const float*)d_in[8];
    const float* Wr_b = (const float*)d_in[10];
    const float* Wu_b = (const float*)d_in[12];
    const float* Wo_b = (const float*)d_in[14];

    char* ws = (char*)d_ws;
    size_t off = 0;
    auto carve = [&](size_t bytes) -> void* {
        void* p = ws + off;
        off += (bytes + 255) & ~(size_t)255;
        return p;
    };
    _Float16* W16[7];
    for (int j = 0; j < 7; ++j) W16[j] = (_Float16*)carve(D_ * D_ * sizeof(_Float16));
    _Float16* X16 = (_Float16*)carve((size_t)MROWS * D_ * sizeof(_Float16));
    float*    Xx  = (float*)   carve((size_t)MROWS * D_ * sizeof(float));
    float*    Xr  = (float*)   carve((size_t)MROWS * D_ * sizeof(float));
    float*    Xu  = (float*)   carve((size_t)MROWS * D_ * sizeof(float));
    _Float16* Hf  = (_Float16*)carve((size_t)MROWS * D_ * sizeof(_Float16));
    _Float16* Hb  = (_Float16*)carve((size_t)MROWS * D_ * sizeof(_Float16));

    for (int j = 0; j < 7; ++j)
        cvt_f32_f16_kernel<<<(D_ * D_ + 255) / 256, 256, 0, stream>>>(Wsrc[j], W16[j], D_ * D_);

    const dim3 blk(512);
    const int GEMM_GRID = 1024;

    // X = inputs @ Wi + Wi_b   (f32 A -> f16 out)
    gemm256_kernel<AM_F32, OM_F16><<<GEMM_GRID, blk, 0, stream>>>(
        inputs, nullptr, W16[0], Wi_b, X16);

    // gate input projections (f16 A via TDM -> f32 out)
    gemm256_kernel<AM_F16, OM_F32><<<GEMM_GRID, blk, 0, stream>>>(
        X16, nullptr, W16[1], Wx_b, Xx);   // x_x = X @ Wx + b
    gemm256_kernel<AM_F16, OM_F32><<<GEMM_GRID, blk, 0, stream>>>(
        X16, nullptr, W16[2], Ux_b, Xr);   // x_r = X @ Ux + b
    gemm256_kernel<AM_F16, OM_F32><<<GEMM_GRID, blk, 0, stream>>>(
        X16, nullptr, W16[3], Rx_b, Xu);   // x_u = X @ Rx + b

    // persistent bidirectional scan (8 WGs: 4 batch tiles x 2 dirs), TDM gate prefetch
    gru_scan_kernel<<<8, blk, SCAN_LDS_BYTES, stream>>>(Xx, Xr, Xu, W16[4], W16[5],
                                                        Wr_b, Wu_b, Hf, Hb);

    // out = (Hf + Hb) @ Wo + Wo_b   (dual-TDM f16 sum A -> f32 out)
    gemm256_kernel<AM_SUM, OM_F32><<<GEMM_GRID, blk, 0, stream>>>(
        Hf, Hb, W16[6], Wo_b, (float*)d_out);
}
